// BertSelfAttention_31963146617403
// MI455X (gfx1250) — matrix operand order repaired and checked by reading, not compiled, not run on hardware
//
#include <hip/hip_runtime.h>
#include <hip/hip_bf16.h>

// ---------------------------------------------------------------------------
// BinaryBERT self-attention for MI455X (gfx1250, wave32, WMMA).
// Store-bandwidth-bound (~830 MB f32 outputs vs ~61 GFLOP): all GEMMs use
// v_wmma_f32_16x16x32_f16, intermediates are fp16 in workspace, every operand
// fragment is a contiguous 16B-chunk load (global_load_b128 / ds_load_b128).
// ---------------------------------------------------------------------------

typedef __attribute__((ext_vector_type(16))) _Float16 v16h;
typedef __attribute__((ext_vector_type(4)))  _Float16 v4h;
typedef __attribute__((ext_vector_type(8)))  float    v8f;

#define NH 12
#define HS 64
#define HID 768
#define BB 16
#define SS 512
#define MTOT (BB * SS)          // 8192 rows of hidden
#define ATT_SCALE 0.125f        // 1/sqrt(64)
#define APAD 776                // 768 + 8 halves: LDS row pad, keeps 16B align

static __device__ __forceinline__ v8f wmma_f16(v16h a, v16h b, v8f c) {
  return __builtin_amdgcn_wmma_f32_16x16x32_f16(
      /*neg_a=*/false, a, /*neg_b=*/false, b,
      /*c_mod=*/(short)0, c, /*reuse_a=*/false, /*reuse_b=*/false);
}

// A fragment: 16(M) x 32(K), source row-major, row stride `ld` halves.
// Lane L: M = L&15, element e holds K = 16*(L>=16) + e.
// One contiguous 32B run per lane.
static __device__ __forceinline__ v16h frag_a(const _Float16* p, int ld, int lane) {
  const _Float16* r = p + (size_t)(lane & 15) * ld + (lane >> 4) * 16;
  v16h a;
#pragma unroll
  for (int e = 0; e < 16; ++e) a[e] = r[e];
  return a;
}

// B fragment: 32(K) x 16(N), source stored [N][K] row-major (row stride `ld`).
// Lane L: N = L&15, element e holds K = 16*(L>=16) + e (one 32B contiguous run).
static __device__ __forceinline__ v16h frag_b(const _Float16* p, int ld, int lane) {
  const _Float16* r = p + (size_t)(lane & 15) * ld + (lane >> 4) * 16;
  v16h b;
#pragma unroll
  for (int e = 0; e < 16; ++e) b[e] = r[e];
  return b;
}

static __device__ __forceinline__ v16h sgn16(v16h x) {
  v16h r;
#pragma unroll
  for (int e = 0; e < 16; ++e) {
    _Float16 v = x[e];
    r[e] = (v > (_Float16)0.f) ? (_Float16)1.f
         : ((v < (_Float16)0.f) ? (_Float16)-1.f : (_Float16)0.f);
  }
  return r;
}

// ---------------------------------------------------------------------------
// Kernel 1: per-row weight scale (mean |w|) + sign(W) in fp16, layout [n][k].
// ---------------------------------------------------------------------------
__global__ void k_prep(const float* __restrict__ Wq, const float* __restrict__ Wk,
                       const float* __restrict__ Wv,
                       _Float16* __restrict__ wsign, float* __restrict__ scales) {
  int row = blockIdx.x;                 // 0..3*768-1
  int mat = row / HID, r = row % HID;
  const float* W = (mat == 0 ? Wq : (mat == 1 ? Wk : Wv)) + (size_t)r * HID;
  _Float16* o = wsign + (size_t)row * HID;
  __shared__ float red[256];
  float s = 0.f;
  for (int k = threadIdx.x; k < HID; k += 256) {
    float w = W[k];
    s += fabsf(w);
    o[k] = (_Float16)((w > 0.f) ? 1.f : (w < 0.f ? -1.f : 0.f));
  }
  red[threadIdx.x] = s;
  __syncthreads();
  for (int t = 128; t > 0; t >>= 1) {
    if (threadIdx.x < t) red[threadIdx.x] += red[threadIdx.x + t];
    __syncthreads();
  }
  if (threadIdx.x == 0) scales[row] = red[0] * (1.f / (float)HID);
}

// ---------------------------------------------------------------------------
// Kernel 2: hidden f32 -> fp16, 4 elements/thread (b128 in, b64 out).
// ---------------------------------------------------------------------------
__global__ void k_cvt(const float4* __restrict__ x, _Float16* __restrict__ y, int n4) {
  int i = blockIdx.x * 256 + threadIdx.x;
  if (i < n4) {
    float4 v = x[i];
    v4h o = { (_Float16)v.x, (_Float16)v.y, (_Float16)v.z, (_Float16)v.w };
    *(v4h*)(y + (size_t)i * 4) = o;
  }
}

// ---------------------------------------------------------------------------
// Kernel 3: QKV projection GEMM.  C[8192, 3*768] = h16 @ signW^T; epilogue
// applies per-column scale & bias, stores fp16 as qkv16[mat][b][h][s][d].
// The 16x768 A-tile is staged once in LDS (shared by all 8 waves); B streams
// from global with an unguarded speculative prefetch.  24 WMMA K-steps/wave.
// ---------------------------------------------------------------------------
__global__ void k_qkv(const _Float16* __restrict__ h16,
                      const _Float16* __restrict__ wsign,
                      const float* __restrict__ scales,
                      const float* __restrict__ bq, const float* __restrict__ bk,
                      const float* __restrict__ bv,
                      _Float16* __restrict__ qkv16) {
  __shared__ _Float16 As[16 * APAD];    // 24.8 KB
  int mtile = blockIdx.x;               // 0..511
  int grp   = blockIdx.y;               // 0..17  (6 groups per matrix)
  int wave  = threadIdx.x >> 5;
  int lane  = threadIdx.x & 31;
  int mat   = grp / 6;
  int ntile = (grp % 6) * 8 + wave;     // 0..47

  // Cooperative A-tile load: 16 rows x 768 halves = 6144 dwords, coalesced.
  {
    const uint32_t* src = (const uint32_t*)(h16 + (size_t)mtile * 16 * HID);
    for (int i = threadIdx.x; i < 16 * (HID / 2); i += 256) {
      int row = i / (HID / 2), col = i % (HID / 2);
      *(uint32_t*)&As[row * APAD + col * 2] = src[i];
    }
  }
  __syncthreads();

  const _Float16* B0 = wsign + ((size_t)mat * HID + (size_t)ntile * 16) * HID;

  v8f c = {};
#pragma unroll 4
  for (int k = 0; k < HID; k += 32) {
    __builtin_prefetch((const void*)(B0 + (size_t)(lane & 15) * HID + k + 256), 0, 3);
    v16h a = frag_a(As + k, APAD, lane);
    v16h b = frag_b(B0 + k, HID, lane);
    c = wmma_f16(a, b, c);
  }

  int n  = ntile * 16 + (lane & 15);
  float sc = scales[mat * HID + n];
  const float* bias = (mat == 0 ? bq : (mat == 1 ? bk : bv));
  float bia = bias[n];
  int h = n >> 6, d = n & 63;
  int hi8 = (lane >> 4) * 8;
#pragma unroll
  for (int j = 0; j < 8; ++j) {
    int m = mtile * 16 + j + hi8;
    int b = m >> 9, s = m & 511;
    size_t idx = ((((size_t)mat * BB + b) * NH + h) * SS + s) * HS + d;
    qkv16[idx] = (_Float16)(c[j] * sc + bia);
  }
}

// ---------------------------------------------------------------------------
// Kernel 4: sign(v) transposed to [b,h,d,s] fp16 via LDS 64x64 tile transpose
// (padded rows), so context-GEMM B-fragments become contiguous b128 loads.
// Grid (8 s-tiles, NH, BB), 256 threads.
// ---------------------------------------------------------------------------
__global__ void k_vsignT(const _Float16* __restrict__ qkv16,
                         _Float16* __restrict__ vsT) {
  __shared__ _Float16 t[64 * 72];       // 9.2 KB, +8 pad vs 64 banks
  int st = blockIdx.x, h = blockIdx.y, b = blockIdx.z;
  const _Float16* vp = qkv16 + (((size_t)2 * BB + b) * NH + h) * SS * HS
                             + (size_t)st * 64 * HS;
  // 64 s-rows x 64 d = 4096 contiguous halves: fully coalesced load.
  for (int i = threadIdx.x; i < 2048; i += 256) {
    uint32_t w = ((const uint32_t*)vp)[i];
    int s = (2 * i) / 64, d = (2 * i) % 64;
    *(uint32_t*)&t[s * 72 + d] = w;
  }
  __syncthreads();
  _Float16* op = vsT + (((size_t)b * NH + h) * HS) * SS + (size_t)st * 64;
  int d  = threadIdx.x >> 2;            // 0..63
  int s0 = (threadIdx.x & 3) * 16;      // 16 contiguous s per thread (32B store)
#pragma unroll
  for (int i = 0; i < 16; ++i) {
    _Float16 x = t[(s0 + i) * 72 + d];
    op[(size_t)d * SS + s0 + i] =
        (x > (_Float16)0.f) ? (_Float16)1.f
      : ((x < (_Float16)0.f) ? (_Float16)-1.f : (_Float16)0.f);
  }
}

// ---------------------------------------------------------------------------
// Kernel 5: per (b,h,qtile) block computes all four 16x512 score strips.
// q/k/v/sign(q) A-fragments live in registers across the whole k loop;
// 8 waves stripe the 32 k-tiles; 8 WMMAs per k-tile.
// Also emits probs = (sign(attn)+1)/2 in fp16 for the context GEMM.
// ---------------------------------------------------------------------------
__global__ void k_scores(const _Float16* __restrict__ qkv16,
                         const float* __restrict__ mask,
                         float* __restrict__ attn_out, float* __restrict__ vs_out,
                         float* __restrict__ qs_out, float* __restrict__ ks_out,
                         _Float16* __restrict__ probs) {
  int qt = blockIdx.x, h = blockIdx.y, b = blockIdx.z;
  int wave = threadIdx.x >> 5, lane = threadIdx.x & 31;

  const _Float16* qp = qkv16 + (((size_t)0 * BB + b) * NH + h) * SS * HS;
  const _Float16* kp = qkv16 + (((size_t)1 * BB + b) * NH + h) * SS * HS;
  const _Float16* vp = qkv16 + (((size_t)2 * BB + b) * NH + h) * SS * HS;

  const _Float16* qrow = qp + (size_t)qt * 16 * HS;
  const _Float16* krow = kp + (size_t)qt * 16 * HS;
  const _Float16* vrow = vp + (size_t)qt * 16 * HS;

  v16h aq0 = frag_a(qrow, HS, lane),      aq1 = frag_a(qrow + 32, HS, lane);
  v16h ak0 = frag_a(krow, HS, lane),      ak1 = frag_a(krow + 32, HS, lane);
  v16h av0 = frag_a(vrow, HS, lane),      av1 = frag_a(vrow + 32, HS, lane);
  v16h aqs0 = sgn16(aq0),                 aqs1 = sgn16(aq1);

  size_t so = (((size_t)b * NH + h) * SS + (size_t)qt * 16) * SS;
  int col0 = lane & 15;
  int hi8  = (lane >> 4) * 8;

  for (int kt = wave; kt < 32; kt += 8) {
    const _Float16* qk = qp + (size_t)kt * 16 * HS;
    const _Float16* kk = kp + (size_t)kt * 16 * HS;
    const _Float16* vk = vp + (size_t)kt * 16 * HS;

    v16h bqf0 = frag_b(qk, HS, lane), bqf1 = frag_b(qk + 32, HS, lane);
    v16h bkf0 = frag_b(kk, HS, lane), bkf1 = frag_b(kk + 32, HS, lane);
    v16h bvf0 = frag_b(vk, HS, lane), bvf1 = frag_b(vk + 32, HS, lane);

    v8f cqq = {}, ckk = {}, cvv = {}, cat = {};
    cqq = wmma_f16(aq0, bqf0, cqq);  cqq = wmma_f16(aq1, bqf1, cqq);
    ckk = wmma_f16(ak0, bkf0, ckk);  ckk = wmma_f16(ak1, bkf1, ckk);
    cvv = wmma_f16(av0, bvf0, cvv);  cvv = wmma_f16(av1, bvf1, cvv);
    cat = wmma_f16(aqs0, sgn16(bkf0), cat);
    cat = wmma_f16(aqs1, sgn16(bkf1), cat);

    int col = kt * 16 + col0;
    float mval = mask[b * SS + col];
#pragma unroll
    for (int j = 0; j < 8; ++j) {
      size_t o = so + (size_t)(j + hi8) * SS + col;
      qs_out[o] = cqq[j] * ATT_SCALE;
      ks_out[o] = ckk[j] * ATT_SCALE;
      vs_out[o] = cvv[j] * ATT_SCALE;
      float av = cat[j] * ATT_SCALE + mval;
      attn_out[o] = av;
      probs[o] = (_Float16)((av > 0.f) ? 1.f : (av < 0.f ? 0.f : 0.5f));
    }
  }
}

// ---------------------------------------------------------------------------
// Kernel 6: context = probs[512x512] @ sign(v)[512x64] per (b,h), written as
// [B,S,H].  A from probs (contiguous), B from vsT (contiguous) -> all b128.
// Grid (32 qtiles, 12, 16), 4 waves/block, one 16-wide d-tile each, 16 WMMAs.
// ---------------------------------------------------------------------------
__global__ void k_context(const _Float16* __restrict__ probs,
                          const _Float16* __restrict__ vsT,
                          float* __restrict__ ctx_out) {
  int qt = blockIdx.x, h = blockIdx.y, b = blockIdx.z;
  int wave = threadIdx.x >> 5, lane = threadIdx.x & 31;

  const _Float16* pr = probs + (((size_t)b * NH + h) * SS + (size_t)qt * 16) * SS;
  const _Float16* vt = vsT + (((size_t)b * NH + h) * HS + (size_t)wave * 16) * SS;

  v8f c = {};
  for (int k0 = 0; k0 < SS; k0 += 32) {
    v16h a  = frag_a(pr + k0, SS, lane);   // probs rows, contiguous in k
    v16h bf = frag_b(vt + k0, SS, lane);   // sign(v)^T rows, contiguous in k
    c = wmma_f16(a, bf, c);
  }
  int d   = wave * 16 + (lane & 15);
  int hi8 = (lane >> 4) * 8;
#pragma unroll
  for (int j = 0; j < 8; ++j) {
    int s = qt * 16 + j + hi8;
    ctx_out[((size_t)b * SS + s) * HID + h * HS + d] = c[j];
  }
}

// ---------------------------------------------------------------------------
extern "C" void kernel_launch(void* const* d_in, const int* in_sizes, int n_in,
                              void* d_out, int out_size, void* d_ws, size_t ws_size,
                              hipStream_t stream) {
  const float* hidden = (const float*)d_in[0];   // [16,512,768]
  const float* amask  = (const float*)d_in[1];   // [16,1,1,512]
  const float* Wq     = (const float*)d_in[2];
  const float* bq     = (const float*)d_in[3];
  const float* Wk     = (const float*)d_in[4];
  const float* bk     = (const float*)d_in[5];
  const float* Wv     = (const float*)d_in[6];
  const float* bv     = (const float*)d_in[7];

  // Outputs, concatenated flat in return order.
  float* out  = (float*)d_out;
  const size_t ctx_n   = (size_t)BB * SS * HID;        // 6,291,456
  const size_t score_n = (size_t)BB * NH * SS * SS;    // 50,331,648
  float* ctx  = out;
  float* attn = ctx + ctx_n;
  float* vals = attn + score_n;
  float* qs   = vals + score_n;
  float* ks   = qs + score_n;

  // Workspace carve-up (fp16 intermediates).
  _Float16* h16    = (_Float16*)d_ws;                        // 8192*768
  _Float16* qkv16  = h16 + (size_t)MTOT * HID;               // 3*8192*768
  _Float16* wsign  = qkv16 + (size_t)3 * MTOT * HID;         // 3*768*768
  float*    scales = (float*)(wsign + (size_t)3 * HID * HID);// 3*768
  _Float16* probs  = (_Float16*)(scales + 3 * HID);          // 16*12*512*512
  _Float16* vsT    = probs + (size_t)BB * NH * SS * SS;      // 16*12*64*512

  // 1) quantize weights
  k_prep<<<dim3(3 * HID), dim3(256), 0, stream>>>(Wq, Wk, Wv, wsign, scales);

  // 2) hidden -> fp16
  {
    int n4 = (MTOT * HID) / 4;
    k_cvt<<<dim3((n4 + 255) / 256), dim3(256), 0, stream>>>(
        (const float4*)hidden, h16, n4);
  }

  // 3) QKV projection GEMM (WMMA, LDS-staged A)
  k_qkv<<<dim3(MTOT / 16, 18), dim3(256), 0, stream>>>(h16, wsign, scales,
                                                       bq, bk, bv, qkv16);

  // 4) sign(v)^T for the context GEMM
  k_vsignT<<<dim3(SS / 64, NH, BB), dim3(256), 0, stream>>>(qkv16, vsT);

  // 5) score matrices + probs (WMMA)
  k_scores<<<dim3(SS / 16, NH, BB), dim3(256), 0, stream>>>(
      qkv16, amask, attn, vals, qs, ks, probs);

  // 6) context GEMM (WMMA)
  k_context<<<dim3(SS / 16, NH, BB), dim3(128), 0, stream>>>(probs, vsT, ctx);
}
